// DGCNN_69947837382934
// MI455X (gfx1250) — matrix operand order repaired and checked by reading, not compiled, not run on hardware
//
#include <hip/hip_runtime.h>

typedef __attribute__((ext_vector_type(16))) _Float16 v16h;
typedef __attribute__((ext_vector_type(8)))  float    v8f;

#define NNODES 65536
#define BGRAPH 256
#define NPG    256
#define NEDGE  (NNODES * 16)
#define KTOP   30
#define DLAT   97      // concat latent dim
#define SPLD   128     // padded sortpool row stride (zero-filled 97..127)
#define C1C    16
#define C2C    32
#define KW2C   5
#define L2LEN  11      // conv2 output length
#define DENSE  352

// ---------------------------------------------------------------------------
// Prepack weight [din x dout] (or transposed [dout x din] for conv1 filters)
// into WMMA B-fragment tiles: tile(kt,nt) -> 32 lanes x 16 halves.
// b[j] (lane) = B[kt*32 + (j<8 ? 8*(lane/16) + j : 16 + 8*(lane/16) + j-8)][nt*16 + lane%16]
// ---------------------------------------------------------------------------
__global__ void k_prepack(const float* __restrict__ W, int din, int dout,
                          int nt_tiles, int transposed, _Float16* __restrict__ dst)
{
    int tile = blockIdx.x;
    int kt = tile / nt_tiles, nt = tile % nt_tiles;
    int lane = threadIdx.x >> 4;     // 0..31
    int j    = threadIdx.x & 15;     // 0..15
    int k = kt * 32 + ((j < 8) ? (((lane >> 4) << 3) + j) : (16 + ((lane >> 4) << 3) + (j - 8)));
    int n = nt * 16 + (lane & 15);
    float v = 0.0f;
    if (k < din && n < dout)
        v = transposed ? W[(size_t)n * din + k] : W[(size_t)k * dout + n];
    dst[(size_t)tile * 512 + lane * 16 + j] = (_Float16)v;
}

// pooled = h  (float4 copy)
__global__ void k_pool_init(const float* __restrict__ h, float* __restrict__ pooled, int total4)
{
    int i = blockIdx.x * blockDim.x + threadIdx.x;
    if (i < total4)
        reinterpret_cast<float4*>(pooled)[i] = reinterpret_cast<const float4*>(h)[i];
}

// pooled[dst] += h[src]   (thread per edge x 4-channel group)
__global__ void k_scatter(const float* __restrict__ h, const int* __restrict__ src,
                          const int* __restrict__ dst, float* __restrict__ pooled,
                          int din4, int total)
{
    int i = blockIdx.x * blockDim.x + threadIdx.x;
    if (i >= total) return;
    int e  = i / din4;
    int c4 = i - e * din4;
    int s = src[e], d = dst[e];
    float4 v = reinterpret_cast<const float4*>(h)[(size_t)s * din4 + c4];
    float* pd = pooled + ((size_t)d * din4 + c4) * 4;
    atomicAdd(pd + 0, v.x);
    atomicAdd(pd + 1, v.y);
    atomicAdd(pd + 2, v.z);
    atomicAdd(pd + 3, v.w);
}

// Per-lane A fragment gather: 16-bit A 16x32 layout (ISA 7.12.2), f32 -> f16.
__device__ __forceinline__ v16h load_a_frag(const float* __restrict__ arow,
                                            int kbase, int half)
{
    v16h a;
#pragma unroll
    for (int v = 0; v < 8; ++v) {
        int kk = kbase + ((v < 4) ? (half * 8 + 2 * v) : (16 + half * 8 + 2 * (v - 4)));
        float2 f = *reinterpret_cast<const float2*>(arow + kk);
        a[2 * v]     = (_Float16)f.x;
        a[2 * v + 1] = (_Float16)f.y;
    }
    return a;
}

// ---------------------------------------------------------------------------
// GNN layer GEMM: out = tanh((pooled @ W + b) / (deg+1))
// One wave per 16-row x 16-col output tile. Writes hnext (if non-null) and
// the Z concat slice at column offset zoff.
// ---------------------------------------------------------------------------
__global__ void __launch_bounds__(256) k_gnn_gemm(
    const float* __restrict__ A, int lda, int kt_tiles,
    const _Float16* __restrict__ pb, int dout, int nt_tiles,
    const float* __restrict__ bias, const int* __restrict__ degs,
    float* __restrict__ hnext, int ldh, float* __restrict__ Z, int zoff)
{
    int wave = (blockIdx.x * blockDim.x + threadIdx.x) >> 5;
    int lane = threadIdx.x & 31;
    int rowTile = wave / nt_tiles;
    int nt      = wave % nt_tiles;
    int rowBase = rowTile << 4;
    int m = lane & 15, half = lane >> 4;
    const float* arow = A + (size_t)(rowBase + m) * lda;

    v8f c = {};
    for (int kt = 0; kt < kt_tiles; ++kt) {
        v16h a = load_a_frag(arow, kt * 32, half);
        v16h b = *reinterpret_cast<const v16h*>(pb + ((size_t)(kt * nt_tiles + nt) << 9) + (lane << 4));
        c = __builtin_amdgcn_wmma_f32_16x16x32_f16(false, a, false, b, (short)0, c, false, false);
    }

    int col = (nt << 4) + (lane & 15);
    bool colOK = col < dout;
    float bc = colOK ? bias[col] : 0.0f;
#pragma unroll
    for (int v = 0; v < 8; ++v) {
        int row = rowBase + (half << 3) + v;
        float deg1 = (float)degs[row] + 1.0f;
        float val = tanhf((c[v] + bc) / deg1);
        if (colOK) {
            if (hnext) hnext[(size_t)row * ldh + col] = val;
            Z[(size_t)row * DLAT + zoff + col] = val;
        }
    }
}

// ---------------------------------------------------------------------------
// SortPooling: per graph, stable top-K rank on last latent channel, gather
// rows into sp [B][K][SPLD] (cols 97..127 zeroed for conv1 WMMA padding).
// ---------------------------------------------------------------------------
__global__ void k_sortpool(const float* __restrict__ Z, float* __restrict__ sp)
{
    __shared__ float vals[NPG];
    int g = blockIdx.x, t = threadIdx.x;
    int node = g * NPG + t;
    float v = Z[(size_t)node * DLAT + (DLAT - 1)];
    vals[t] = v;
    __syncthreads();
    int rank = 0;
    for (int j = 0; j < NPG; ++j) {
        float vj = vals[j];
        rank += (vj > v) || (vj == v && j < t);
    }
    if (rank < KTOP) {
        float* dstp = sp + ((size_t)g * KTOP + rank) * SPLD;
        const float* srcp = Z + (size_t)node * DLAT;
        for (int d = 0; d < DLAT; ++d) dstp[d] = srcp[d];
        for (int d = DLAT; d < SPLD; ++d) dstp[d] = 0.0f;
    }
}

// ---------------------------------------------------------------------------
// conv1 (kernel width = D, stride = D) as GEMM [B*K x 128(pad)] @ [128 x 16]
// + bias + ReLU; store NCH-transposed into c1out[g][c1][kpos].
// ---------------------------------------------------------------------------
__global__ void __launch_bounds__(256) k_conv1(
    const float* __restrict__ sp, const _Float16* __restrict__ pb,
    const float* __restrict__ bias, float* __restrict__ c1out)
{
    int wave = (blockIdx.x * blockDim.x + threadIdx.x) >> 5;
    int lane = threadIdx.x & 31;
    int rowBase = wave << 4;
    int m = lane & 15, half = lane >> 4;
    const float* arow = sp + (size_t)(rowBase + m) * SPLD;

    v8f c = {};
#pragma unroll
    for (int kt = 0; kt < 4; ++kt) {
        v16h a = load_a_frag(arow, kt * 32, half);
        v16h b = *reinterpret_cast<const v16h*>(pb + ((size_t)kt << 9) + (lane << 4));
        c = __builtin_amdgcn_wmma_f32_16x16x32_f16(false, a, false, b, (short)0, c, false, false);
    }

    int col = lane & 15;  // c1 channel (all 16 valid)
    float bc = bias[col];
#pragma unroll
    for (int v = 0; v < 8; ++v) {
        int row = rowBase + (half << 3) + v;      // row = g*KTOP + kpos
        int g = row / KTOP, kpos = row - g * KTOP;
        float val = fmaxf(c[v] + bc, 0.0f);
        c1out[((size_t)g * C1C + col) * KTOP + kpos] = val;
    }
}

// Fused maxpool(2,2) + conv2(width 5) + ReLU.
__global__ void k_conv2(const float* __restrict__ c1out, const float* __restrict__ w,
                        const float* __restrict__ b, float* __restrict__ c2out)
{
    int i = blockIdx.x * blockDim.x + threadIdx.x;
    if (i >= BGRAPH * C2C * L2LEN) return;
    int p  = i % L2LEN;
    int c2 = (i / L2LEN) % C2C;
    int g  = i / (L2LEN * C2C);
    float acc = b[c2];
    for (int ci = 0; ci < C1C; ++ci) {
        const float* row = c1out + ((size_t)g * C1C + ci) * KTOP;
#pragma unroll
        for (int kw = 0; kw < KW2C; ++kw) {
            int j = p + kw;
            float mv = fmaxf(row[2 * j], row[2 * j + 1]);
            acc += mv * w[((size_t)c2 * C1C + ci) * KW2C + kw];
        }
    }
    c2out[((size_t)g * C2C + c2) * L2LEN + p] = fmaxf(acc, 0.0f);
}

// Final dense [B,352]@[352,2] + bias + relu(relu(x)).
__global__ void k_dense(const float* __restrict__ x, const float* __restrict__ w,
                        const float* __restrict__ b, float* __restrict__ out)
{
    int i = blockIdx.x * blockDim.x + threadIdx.x;
    if (i >= BGRAPH * 2) return;
    int o = i & 1, g = i >> 1;
    float acc = b[o];
    const float* xr = x + (size_t)g * DENSE;
    for (int d = 0; d < DENSE; ++d) acc += xr[d] * w[d * 2 + o];
    out[i] = fmaxf(acc, 0.0f);
}

extern "C" void kernel_launch(void* const* d_in, const int* in_sizes, int n_in,
                              void* d_out, int out_size, void* d_ws, size_t ws_size,
                              hipStream_t stream)
{
    // setup_inputs() dict order
    const float* node_feat = (const float*)d_in[0];
    const int*   src       = (const int*)d_in[1];
    const int*   dst       = (const int*)d_in[2];
    const int*   degs      = (const int*)d_in[3];
    const float* Wm[4]   = { (const float*)d_in[4], (const float*)d_in[6],
                             (const float*)d_in[8], (const float*)d_in[10] };
    const float* bm[4]   = { (const float*)d_in[5], (const float*)d_in[7],
                             (const float*)d_in[9], (const float*)d_in[11] };
    const float* conv1_w = (const float*)d_in[12];
    const float* conv1_b = (const float*)d_in[13];
    const float* conv2_w = (const float*)d_in[14];
    const float* conv2_b = (const float*)d_in[15];
    const float* out_w   = (const float*)d_in[16];
    const float* out_b   = (const float*)d_in[17];
    float* out = (float*)d_out;

    // ---- workspace carve-up ----
    char* ws = (char*)d_ws;
    size_t off = 0;
    float* pooled = (float*)(ws + off); off += (size_t)NNODES * 128 * 4;   // 32 MB
    float* h      = (float*)(ws + off); off += (size_t)NNODES * 32 * 4;    //  8 MB
    float* Z      = (float*)(ws + off); off += (size_t)NNODES * DLAT * 4;  // 25 MB
    float* sp     = (float*)(ws + off); off += (size_t)BGRAPH * KTOP * SPLD * 4;
    float* c1out  = (float*)(ws + off); off += (size_t)BGRAPH * C1C * KTOP * 4;
    float* c2out  = (float*)(ws + off); off += (size_t)BGRAPH * C2C * L2LEN * 4;
    _Float16* pb  = (_Float16*)(ws + off); off += (size_t)32 * 512 * 2;
    (void)ws_size; (void)in_sizes; (void)n_in; (void)out_size;

    _Float16* pb0  = pb;            // 8 tiles (128x32)
    _Float16* pb1  = pb + 8 * 512;  // 2 tiles (32x32)
    _Float16* pb2  = pb + 10 * 512; // 2 tiles
    _Float16* pb3  = pb + 12 * 512; // 1 tile  (32x1 padded)
    _Float16* pbc1 = pb + 13 * 512; // 4 tiles (97x16 padded to 128)

    // ---- prepack weights into WMMA B-fragment layout ----
    k_prepack<<<8, 512, 0, stream>>>(Wm[0], 128, 32, 2, 0, pb0);
    k_prepack<<<2, 512, 0, stream>>>(Wm[1],  32, 32, 2, 0, pb1);
    k_prepack<<<2, 512, 0, stream>>>(Wm[2],  32, 32, 2, 0, pb2);
    k_prepack<<<1, 512, 0, stream>>>(Wm[3],  32,  1, 1, 0, pb3);
    k_prepack<<<4, 512, 0, stream>>>(conv1_w, 97, 16, 1, 1, pbc1);

    // ---- 4 GNN layers ----
    const int dins[4]  = {128, 32, 32, 32};
    const int douts[4] = {32, 32, 32, 1};
    const float* Ain[4] = {node_feat, h, h, h};
    const _Float16* pbs[4] = {pb0, pb1, pb2, pb3};
    const int zoffs[4] = {0, 32, 64, 96};

    for (int L = 0; L < 4; ++L) {
        int din = dins[L], dout = douts[L];
        int nt = (dout + 15) / 16;
        int din4 = din / 4;

        int total4 = NNODES * din4;
        k_pool_init<<<total4 / 256, 256, 0, stream>>>(Ain[L], pooled, total4);

        int stot = NEDGE * din4;
        k_scatter<<<(stot + 255) / 256, 256, 0, stream>>>(Ain[L], src, dst, pooled, din4, stot);

        int jobs = (NNODES / 16) * nt;             // exact multiple of 8
        k_gnn_gemm<<<jobs / 8, 256, 0, stream>>>(
            pooled, din, din / 32, pbs[L], dout, nt, bm[L], degs,
            (L < 3) ? h : nullptr, 32, Z, zoffs[L]);
    }

    // ---- sortpool + conv stack ----
    k_sortpool<<<BGRAPH, NPG, 0, stream>>>(Z, sp);

    int c1jobs = (BGRAPH * KTOP) / 16;             // 480 waves
    k_conv1<<<c1jobs / 8, 256, 0, stream>>>(sp, pbc1, conv1_b, c1out);

    int c2n = BGRAPH * C2C * L2LEN;
    k_conv2<<<(c2n + 255) / 256, 256, 0, stream>>>(c1out, conv2_w, conv2_b, c2out);

    k_dense<<<(BGRAPH * 2 + 255) / 256, 256, 0, stream>>>(c2out, out_w, out_b, out);
}
